// FiLMGNNBackbone_88940182765942
// MI455X (gfx1250) — matrix operand order repaired and checked by reading, not compiled, not run on hardware
//
#include <hip/hip_runtime.h>
#include <hip/hip_bf16.h>

typedef unsigned short u16;
typedef __attribute__((ext_vector_type(16))) __bf16 v16bf;
typedef __attribute__((ext_vector_type(8)))  float  v8f;

constexpr int Bb = 2;        // batch
constexpr int Nn = 10000;    // nodes
constexpr int Ee = 320000;   // edges
constexpr int Dd = 128;      // hidden
constexpr int NT = (Nn + 63) / 64;   // 157 node tiles

// ---------- helpers ----------
__device__ __forceinline__ u16 f2bf(float x) {
  __bf16 b = (__bf16)x;               // native v_cvt_bf16_f32 (RNE)
  union { __bf16 b; u16 s; } u; u.b = b;
  return u.s;
}
__device__ __forceinline__ unsigned pack2bf(float lo, float hi) {
  return (unsigned)f2bf(lo) | ((unsigned)f2bf(hi) << 16);
}
__device__ __forceinline__ float silu(float x) { return x / (1.f + __expf(-x)); }

// A fragment: 16x32 bf16 tile from LDS row-major [rows][ldk].
// Per ISA: lane half h, VGPR 0..3 hold K = k0+8h .. k0+8h+7 (contiguous),
// VGPR 4..7 hold K = k0+16+8h .. +7 (contiguous) -> two 16B vector loads.
__device__ __forceinline__ v16bf lds_a(const u16* t, int ldk, int rbase, int k0, int lane) {
  int half = lane >> 4, m = lane & 15;
  const u16* row = t + (rbase + m) * ldk + k0 + 8 * half;
  union { v16bf v; uint4 q[2]; } u;
  u.q[0] = *(const uint4*)(row);
  u.q[1] = *(const uint4*)(row + 16);
  return u.v;
}
// B fragment from fragment-swizzled weights: [frag][lane][16 bf16] contiguous.
__device__ __forceinline__ v16bf lds_bsw(const u16* wsw, int frag, int lane) {
  const u16* p = wsw + ((size_t)frag * 32 + lane) * 16;
  union { v16bf v; uint4 q[2]; } u;
  u.q[0] = ((const uint4*)p)[0];
  u.q[1] = ((const uint4*)p)[1];
  return u.v;
}
__device__ __forceinline__ v8f wmma_bf(v16bf a, v16bf b, v8f c) {
  return __builtin_amdgcn_wmma_f32_16x16x32_bf16(false, a, false, b, (short)0, c, false, false);
}

// ---------- plain f32 -> bf16 conversion (row-major, for edge features) ----------
__global__ void cvt_pad_kernel(const float* __restrict__ src, u16* __restrict__ dst, int n) {
  int i = blockIdx.x * blockDim.x + threadIdx.x;
  if (i < n) dst[i] = f2bf(src[i]);
}

// ---------- weight f32[K][128] -> bf16 fragment-swizzled [K/32][8][32][16] ----------
__global__ void cvt_swz_kernel(const float* __restrict__ src, u16* __restrict__ dst,
                               int K, int Kpad) {
  int i = blockIdx.x * blockDim.x + threadIdx.x;
  if (i >= Kpad * 128) return;
  int j  = i & 15;          // element within lane (K offset)
  int l  = (i >> 4) & 31;   // lane
  int nt = (i >> 9) & 7;    // n tile
  int kk = i >> 12;         // k step
  int k = kk * 32 + ((l >> 4) << 3) + (j & 7) + ((j >> 3) << 4);
  int n = nt * 16 + (l & 15);
  dst[i] = (k < K) ? f2bf(src[k * 128 + n]) : (u16)0;
}

// ---------- g_emb = MLP(global) : tiny, one block per batch ----------
__global__ __launch_bounds__(256) void gemb_kernel(
    const float* __restrict__ gf, const float* __restrict__ g1W, const float* __restrict__ g1b,
    const float* __restrict__ g2W, const float* __restrict__ g2b, float* __restrict__ gemb) {
  __shared__ float t1[128];
  int b = blockIdx.x, tid = threadIdx.x;
  if (tid < 128) {
    float x = gf[b * 2 + 0] * g1W[0 * 128 + tid] + gf[b * 2 + 1] * g1W[1 * 128 + tid] + g1b[tid];
    t1[tid] = silu(x);
  }
  __syncthreads();
  if (tid < 128) {
    float s = g2b[tid];
    for (int k = 0; k < 128; ++k) s += t1[k] * g2W[k * 128 + tid];
    gemb[b * 128 + tid] = s;
  }
}

// ---------- per-layer FiLM vector (2x256), one block per batch ----------
__global__ __launch_bounds__(256) void film_kernel(
    const float* __restrict__ gemb,
    const float* __restrict__ f1W, const float* __restrict__ f1b,
    const float* __restrict__ f2W, const float* __restrict__ f2b,
    float* __restrict__ filmL) {
  __shared__ float hid[256];
  int b = blockIdx.x, tid = threadIdx.x;
  {
    float s = f1b[tid];
    for (int k = 0; k < 128; ++k) s += gemb[b * 128 + k] * f1W[k * 256 + tid];
    hid[tid] = silu(s);
  }
  __syncthreads();
  {
    float s = f2b[tid];
    for (int k = 0; k < 256; ++k) s += hid[k] * f2W[k * 256 + tid];
    filmL[b * 256 + tid] = s;
  }
}

// ---------- node projection: h = nf @ Wproj + b  (64 nodes / block, WMMA) ----------
__global__ __launch_bounds__(256) void proj_kernel(
    const float* __restrict__ nf, const u16* __restrict__ W, const float* __restrict__ bias,
    float* __restrict__ h_out, u16* __restrict__ hbf_out) {
  extern __shared__ char smem[];
  u16* At = (u16*)smem;             // 64 x 64
  u16* Ws = At + 64 * 64;           // swizzled 64x128
  int tid = threadIdx.x, lane = tid & 31, wave = tid >> 5;
  int nbase = blockIdx.x * 64, b = blockIdx.y;

  for (int i = tid; i < 64 * 64; i += 256) {
    int r = i >> 6, c = i & 63, node = nbase + r;
    At[i] = (node < Nn) ? f2bf(nf[((size_t)b * Nn + node) * 64 + c]) : (u16)0;
  }
  for (int i = tid; i < 64 * 128 / 8; i += 256) ((uint4*)Ws)[i] = ((const uint4*)W)[i];
  __syncthreads();

  int rt = wave >> 1, nt0 = (wave & 1) * 4;
  v8f acc[4] = {};
  for (int kk = 0; kk < 2; ++kk) {
    v16bf a = lds_a(At, 64, rt * 16, kk * 32, lane);
#pragma unroll
    for (int t = 0; t < 4; ++t)
      acc[t] = wmma_bf(a, lds_bsw(Ws, kk * 8 + nt0 + t, lane), acc[t]);
  }
  int n0 = lane & 15, half = lane >> 4;
#pragma unroll
  for (int t = 0; t < 4; ++t) {
    int n = (nt0 + t) * 16 + n0;
    float bv = bias[n];
#pragma unroll
    for (int v = 0; v < 8; ++v) {
      int node = nbase + rt * 16 + v + 8 * half;
      if (node < Nn) {
        float y = acc[t][v] + bv;
        size_t o = (size_t)b * Nn * Dd + (size_t)node * Dd + n;
        h_out[o] = y;
        hbf_out[o] = f2bf(y);
      }
    }
  }
}

// ---------- fused edge message MLP + scatter-add (64 edges / block) ----------
__global__ __launch_bounds__(256) void msg_kernel(
    const u16* __restrict__ h_bf, const u16* __restrict__ ef_bf, const int* __restrict__ eidx,
    const u16* __restrict__ W1, const u16* __restrict__ W2,
    const float* __restrict__ b1, const float* __restrict__ b2,
    float* __restrict__ agg) {
  extern __shared__ char smem[];
  u16* At  = (u16*)smem;            // 64 x 288 (K padded 272->288)
  u16* W1s = At + 64 * 288;         // swizzled 288x128
  u16* W2s = W1s + 288 * 128;       // swizzled 128x128
  u16* act = W2s + 128 * 128;       // 64 x 128
  int* srcs = (int*)(act + 64 * 128);  // 64
  int* dsts = srcs + 64;               // 64
  int tid = threadIdx.x, lane = tid & 31, wave = tid >> 5;
  int ebase = blockIdx.x * 64, b = blockIdx.y;
  const u16* hb = h_bf + (size_t)b * Nn * Dd;

  if (tid < 64) { srcs[tid] = eidx[ebase + tid]; dsts[tid] = eidx[Ee + ebase + tid]; }
  __syncthreads();

  // gather A tile: [h[src] | h[dst] | ef | 0-pad]  -- 36 uint4 chunks per row
  for (int i = tid; i < 64 * 36; i += 256) {
    int r = i / 36, c = i % 36;
    uint4 val = make_uint4(0u, 0u, 0u, 0u);
    if (c < 16)       val = *(const uint4*)(hb + (size_t)srcs[r] * Dd + c * 8);
    else if (c < 32)  val = *(const uint4*)(hb + (size_t)dsts[r] * Dd + (c - 16) * 8);
    else if (c < 34)  val = *(const uint4*)(ef_bf + (size_t)(ebase + r) * 16 + (c - 32) * 8);
    *(uint4*)(At + r * 288 + c * 8) = val;
  }
  for (int i = tid; i < (288 * 128) / 8; i += 256) ((uint4*)W1s)[i] = ((const uint4*)W1)[i];
  for (int i = tid; i < (128 * 128) / 8; i += 256) ((uint4*)W2s)[i] = ((const uint4*)W2)[i];
  __syncthreads();

  int rt = wave >> 1, nt0 = (wave & 1) * 4;
  int n0 = lane & 15, half = lane >> 4;

  // GEMM1: (64 x 288) @ (288 x 128)
  v8f acc[4] = {};
  for (int kk = 0; kk < 9; ++kk) {
    v16bf a = lds_a(At, 288, rt * 16, kk * 32, lane);
#pragma unroll
    for (int t = 0; t < 4; ++t)
      acc[t] = wmma_bf(a, lds_bsw(W1s, kk * 8 + nt0 + t, lane), acc[t]);
  }
  // bias + SiLU -> act (bf16)
#pragma unroll
  for (int t = 0; t < 4; ++t) {
    int n = (nt0 + t) * 16 + n0;
    float bv = b1[n];
#pragma unroll
    for (int v = 0; v < 8; ++v) {
      int m = rt * 16 + v + 8 * half;
      act[m * 128 + n] = f2bf(silu(acc[t][v] + bv));
    }
  }
  __syncthreads();

  // GEMM2: (64 x 128) @ (128 x 128)
  v8f acc2[4] = {};
  for (int kk = 0; kk < 4; ++kk) {
    v16bf a = lds_a(act, 128, rt * 16, kk * 32, lane);
#pragma unroll
    for (int t = 0; t < 4; ++t)
      acc2[t] = wmma_bf(a, lds_bsw(W2s, kk * 8 + nt0 + t, lane), acc2[t]);
  }
  // scatter-add to agg[dst]
  float* aggb = agg + (size_t)b * Nn * Dd;
#pragma unroll
  for (int t = 0; t < 4; ++t) {
    int n = (nt0 + t) * 16 + n0;
    float bv = b2[n];
#pragma unroll
    for (int v = 0; v < 8; ++v) {
      int m = rt * 16 + v + 8 * half;
      atomicAdd(aggb + (size_t)dsts[m] * Dd + n, acc2[t][v] + bv);
    }
  }
}

// ---------- fused node update MLP + residual + LayerNorm + FiLM ----------
__global__ __launch_bounds__(256) void upd_kernel(
    const u16* __restrict__ h_bf, const float* __restrict__ agg, const float* __restrict__ h_f32,
    const u16* __restrict__ W1, const u16* __restrict__ W2,
    const float* __restrict__ b1, const float* __restrict__ b2,
    const float* __restrict__ lng, const float* __restrict__ lnb,
    const float* __restrict__ filmL,
    float* __restrict__ h_out, u16* __restrict__ hbf_out) {
  extern __shared__ char smem[];
  u16* At  = (u16*)smem;            // 64 x 256
  u16* W1s = At + 64 * 256;         // swizzled 256x128 (reused as f32 hnew[64][128])
  u16* W2s = W1s + 256 * 128;       // swizzled 128x128
  u16* act = W2s + 128 * 128;       // 64 x 128
  float* hnew = (float*)W1s;
  int tid = threadIdx.x, lane = tid & 31, wave = tid >> 5;
  int nbase = blockIdx.x * 64, b = blockIdx.y;
  const u16*  hb = h_bf + (size_t)b * Nn * Dd;
  const float* ab = agg + (size_t)b * Nn * Dd;

  for (int i = tid; i < 64 * 256; i += 256) {
    int r = i >> 8, c = i & 255, node = nbase + r;
    u16 v = 0;
    if (node < Nn)
      v = (c < 128) ? hb[(size_t)node * Dd + c] : f2bf(ab[(size_t)node * Dd + (c - 128)]);
    At[i] = v;
  }
  for (int i = tid; i < (256 * 128) / 8; i += 256) ((uint4*)W1s)[i] = ((const uint4*)W1)[i];
  for (int i = tid; i < (128 * 128) / 8; i += 256) ((uint4*)W2s)[i] = ((const uint4*)W2)[i];
  __syncthreads();

  int rt = wave >> 1, nt0 = (wave & 1) * 4;
  int n0 = lane & 15, half = lane >> 4;

  v8f acc[4] = {};
  for (int kk = 0; kk < 8; ++kk) {
    v16bf a = lds_a(At, 256, rt * 16, kk * 32, lane);
#pragma unroll
    for (int t = 0; t < 4; ++t)
      acc[t] = wmma_bf(a, lds_bsw(W1s, kk * 8 + nt0 + t, lane), acc[t]);
  }
#pragma unroll
  for (int t = 0; t < 4; ++t) {
    int n = (nt0 + t) * 16 + n0;
    float bv = b1[n];
#pragma unroll
    for (int v = 0; v < 8; ++v) {
      int m = rt * 16 + v + 8 * half;
      act[m * 128 + n] = f2bf(silu(acc[t][v] + bv));
    }
  }
  __syncthreads();   // all GEMM1 reads of W1s done; W1s region now reusable

  v8f acc2[4] = {};
  for (int kk = 0; kk < 4; ++kk) {
    v16bf a = lds_a(act, 128, rt * 16, kk * 32, lane);
#pragma unroll
    for (int t = 0; t < 4; ++t)
      acc2[t] = wmma_bf(a, lds_bsw(W2s, kk * 8 + nt0 + t, lane), acc2[t]);
  }
#pragma unroll
  for (int t = 0; t < 4; ++t) {
    int n = (nt0 + t) * 16 + n0;
    float bv = b2[n];
#pragma unroll
    for (int v = 0; v < 8; ++v) {
      int m = rt * 16 + v + 8 * half;
      hnew[m * 128 + n] = acc2[t][v] + bv;
    }
  }
  __syncthreads();

  // epilogue: residual + LN + FiLM (one thread per node row)
  if (tid < 64) {
    int node = nbase + tid;
    if (node < Nn) {
      const float* ho = h_f32 + (size_t)b * Nn * Dd + (size_t)node * Dd;
      float s = 0.f, s2 = 0.f;
      for (int j = 0; j < 128; ++j) {
        float x = ho[j] + hnew[tid * 128 + j];
        hnew[tid * 128 + j] = x;
        s += x; s2 += x * x;
      }
      float m = s * (1.f / 128.f);
      float var = s2 * (1.f / 128.f) - m * m;
      float inv = rsqrtf(var + 1e-5f);
      const float* fl = filmL + b * 256;
      size_t o = (size_t)b * Nn * Dd + (size_t)node * Dd;
      for (int j = 0; j < 128; j += 2) {
        float y0 = (hnew[tid * 128 + j] - m) * inv * lng[j] + lnb[j];
        y0 = y0 * (1.f + fl[j]) + fl[128 + j];
        float y1 = (hnew[tid * 128 + j + 1] - m) * inv * lng[j + 1] + lnb[j + 1];
        y1 = y1 * (1.f + fl[j + 1]) + fl[128 + j + 1];
        *(float2*)(h_out + o + j) = make_float2(y0, y1);
        *(unsigned*)(hbf_out + o + j) = pack2bf(y0, y1);
      }
    }
  }
}

// ---------- host side ----------
extern "C" void kernel_launch(void* const* d_in, const int* in_sizes, int n_in,
                              void* d_out, int out_size, void* d_ws, size_t ws_size,
                              hipStream_t stream) {
  (void)in_sizes; (void)n_in; (void)out_size;
  const float* nf    = (const float*)d_in[0];
  const int*   eidx  = (const int*)d_in[1];
  const float* ef    = (const float*)d_in[2];
  const float* gf    = (const float*)d_in[3];
  const float* g1W   = (const float*)d_in[4];
  const float* g1b   = (const float*)d_in[5];
  const float* g2W   = (const float*)d_in[6];
  const float* g2b   = (const float*)d_in[7];
  const float* projW = (const float*)d_in[8 + 4 * 14];
  const float* projB = (const float*)d_in[8 + 4 * 14 + 1];

  char* ws = (char*)d_ws;
  size_t off = 0;
  auto alloc = [&](size_t bytes) -> void* {
    void* p = ws + off;
    off += (bytes + 255) & ~((size_t)255);
    return p;
  };
  const size_t HN = (size_t)Bb * Nn * Dd;
  float* h     = (float*)alloc(HN * 4);
  u16*   hbf   = (u16*)  alloc(HN * 2);
  float* agg   = (float*)alloc(HN * 4);
  u16*   efbf  = (u16*)  alloc((size_t)Ee * 16 * 2);
  float* gemb  = (float*)alloc(Bb * 128 * 4);
  float* film  = (float*)alloc(4 * Bb * 256 * 4);
  u16*   projWb = (u16*) alloc(64 * 128 * 2);
  u16 *m1p[4], *m2p[4], *u1p[4], *u2p[4];
  for (int L = 0; L < 4; ++L) {
    m1p[L] = (u16*)alloc(288 * 128 * 2);
    m2p[L] = (u16*)alloc(128 * 128 * 2);
    u1p[L] = (u16*)alloc(256 * 128 * 2);
    u2p[L] = (u16*)alloc(128 * 128 * 2);
  }
  if (off > ws_size) return;

  auto swz = [&](const float* src, u16* dst, int K, int Kpad) {
    int total = Kpad * 128;
    cvt_swz_kernel<<<(total + 255) / 256, 256, 0, stream>>>(src, dst, K, Kpad);
  };

  // conversions: edge features (row-major) + all weights (fragment-swizzled)
  {
    int total = Ee * 16;
    cvt_pad_kernel<<<(total + 255) / 256, 256, 0, stream>>>(ef, efbf, total);
  }
  swz(projW, projWb, 64, 64);
  for (int L = 0; L < 4; ++L) {
    int base = 8 + L * 14;
    swz((const float*)d_in[base + 6],  m1p[L], 272, 288);  // msg1.W (K padded)
    swz((const float*)d_in[base + 8],  m2p[L], 128, 128);  // msg2.W
    swz((const float*)d_in[base + 10], u1p[L], 256, 256);  // upd1.W
    swz((const float*)d_in[base + 12], u2p[L], 128, 128);  // upd2.W
  }

  // global embedding + all FiLM vectors (g_emb is layer-invariant)
  gemb_kernel<<<Bb, 256, 0, stream>>>(gf, g1W, g1b, g2W, g2b, gemb);
  for (int L = 0; L < 4; ++L) {
    int base = 8 + L * 14;
    film_kernel<<<Bb, 256, 0, stream>>>(
        gemb,
        (const float*)d_in[base + 0], (const float*)d_in[base + 1],
        (const float*)d_in[base + 2], (const float*)d_in[base + 3],
        film + L * Bb * 256);
  }

  // node projection
  proj_kernel<<<dim3(NT, Bb), 256, 64 * 64 * 2 + 64 * 128 * 2, stream>>>(
      nf, projWb, projB, h, hbf);

  const size_t msgLds = (size_t)(64 * 288 + 288 * 128 + 128 * 128 + 64 * 128) * 2 + 128 * 4;
  const size_t updLds = (size_t)(64 * 256 + 256 * 128 + 128 * 128 + 64 * 128) * 2;

  for (int L = 0; L < 4; ++L) {
    int base = 8 + L * 14;
    hipMemsetAsync(agg, 0, HN * 4, stream);
    msg_kernel<<<dim3(Ee / 64, Bb), 256, msgLds, stream>>>(
        hbf, efbf, eidx, m1p[L], m2p[L],
        (const float*)d_in[base + 7], (const float*)d_in[base + 9], agg);
    upd_kernel<<<dim3(NT, Bb), 256, updLds, stream>>>(
        hbf, agg, h, u1p[L], u2p[L],
        (const float*)d_in[base + 11], (const float*)d_in[base + 13],
        (const float*)d_in[base + 4], (const float*)d_in[base + 5],
        film + L * Bb * 256,
        (L == 3) ? (float*)d_out : h, hbf);
  }
}
